// GINEncoder_40621800685938
// MI455X (gfx1250) — matrix operand order repaired and checked by reading, not compiled, not run on hardware
//
#include <hip/hip_runtime.h>
#include <hip/hip_bf16.h>

// ---------------------------------------------------------------------------
// GIN encoder for MI455X (gfx1250): bf16 WMMA GEMMs + async-LDS weight staging
// + atomic scatter/pool. Bandwidth-bound op (~0.1-0.2 ms at 23.3 TB/s).
// GEMMs: v_wmma_f32_16x16x32_bf16, weights staged once per workgroup into LDS
// via global_load_async_to_lds_b128 (ASYNCcnt), B-fragments read as ds_load_b128.
// ---------------------------------------------------------------------------

constexpr int N_NODES  = 100000;
constexpr int N_EDGES  = 800000;
constexpr int IN_DIM   = 128;
constexpr int HID      = 256;
constexpr int N_GRAPHS = 256;
constexpr float BN_EPS = 1e-5f;

typedef __attribute__((ext_vector_type(16))) __bf16 v16bf;
typedef __attribute__((ext_vector_type(8)))  float  v8f;

typedef __attribute__((address_space(3))) unsigned short lds_us;

struct Frag32 { uint4 lo, hi; };   // 32 bytes == one v16bf fragment

__device__ __forceinline__ unsigned short f2bf(float x) {
  // round-to-nearest-even fp32 -> bf16
  unsigned int u = __float_as_uint(x);
  u += 0x7FFFu + ((u >> 16) & 1u);
  return (unsigned short)(u >> 16);
}

// ---------------------------------------------------------------------------
// small utility kernels
// ---------------------------------------------------------------------------
__global__ void zero_kernel(float* p, int n) {
  int i = blockIdx.x * blockDim.x + threadIdx.x;
  if (i < n) p[i] = 0.0f;
}

__global__ void copy_kernel(const float* __restrict__ a, float* __restrict__ o, int n) {
  int i = blockIdx.x * blockDim.x + threadIdx.x;
  if (i < n) o[i] = a[i];
}

__global__ void f2bf_kernel(const float* __restrict__ a, unsigned short* __restrict__ o, int n) {
  int i = blockIdx.x * blockDim.x + threadIdx.x;
  if (i < n) o[i] = f2bf(a[i]);
}

// W (K x HID row-major, fp32) -> Wt (HID x K row-major, bf16): per-lane WMMA
// B-fragments become two contiguous 16B loads.
template<int K>
__global__ void wconvert_kernel(const float* __restrict__ W, unsigned short* __restrict__ Wt) {
  int i = blockIdx.x * blockDim.x + threadIdx.x;   // i = k*HID + n (coalesced read)
  if (i >= K * HID) return;
  int k = i / HID, n = i % HID;
  Wt[(size_t)n * K + k] = f2bf(W[i]);
}

// agg[dst] += h[src]  (agg pre-initialized with h itself => (1+eps)*x + sum)
template<int D>
__global__ void scatter_kernel(const float* __restrict__ h, const int* __restrict__ ei,
                               float* __restrict__ agg) {
  long long tid = (long long)blockIdx.x * blockDim.x + threadIdx.x;
  if (tid >= (long long)N_EDGES * D) return;
  int e = (int)(tid / D);
  int f = (int)(tid % D);
  int s = ei[e];                 // edge_index[0][e]
  int d = ei[N_EDGES + e];       // edge_index[1][e]
  atomicAdd(&agg[(size_t)d * D + f], h[(size_t)s * D + f]);
}

// ---------------------------------------------------------------------------
// bf16 WMMA GEMM:  out[M x 256] = act(A[M x K](bf16) @ W[K x 256] + bias)
// Wt is the weight pre-transposed to N x K (bf16) and staged whole into LDS
// once per workgroup with gfx1250 async global->LDS DMA. One wave computes a
// 16 x 256 output stripe; 8 waves per block share the LDS weight copy.
// A fragment layout (16-bit A, 16x32): lanes 0-15 row M=0..15, K 0-7 & 16-23;
// lanes 16-31 K 8-15 & 24-31 -> two 16B loads per lane.
// B fragment (32x16): lane holds one column, K 0-7 & 16-23 (lanes 16-31: 8-15 & 24-31) from LDS.
// ---------------------------------------------------------------------------
template<int K, bool RELU, bool OUT_BF16>
__launch_bounds__(256)
__global__ void gemm_kernel(const unsigned short* __restrict__ A,
                            const unsigned short* __restrict__ Wt,
                            const float* __restrict__ bias,
                            void* __restrict__ out, int M) {
  extern __shared__ unsigned short lds_w[];     // HID * K bf16 (64KB or 128KB)
  constexpr int KS     = K / 32;
  constexpr int CHUNKS = HID * K / 8;           // 16-byte chunks to stage

  const int lane  = threadIdx.x & 31;
  const int wave  = threadIdx.x >> 5;
  const int mtile = blockIdx.x * 8 + wave;
  const bool active = (mtile * 16 < M);         // wave-uniform predicate
  const int half = lane >> 4;                   // 0 or 1
  const int low  = lane & 15;

  // ---- A fragments: issue global loads first so they fly during staging ----
  v16bf afrag[KS];
  if (active) {
    const size_t arow = (size_t)(mtile * 16 + low) * K;
#pragma unroll
    for (int ks = 0; ks < KS; ++ks) {
      const unsigned short* ap = A + arow + ks * 32 + half * 8;
      Frag32 f;
      f.lo = *(const uint4*)(ap);         // K chunk0 (8 bf16)
      f.hi = *(const uint4*)(ap + 16);    // K chunk0+16 (8 bf16)
      afrag[ks] = __builtin_bit_cast(v16bf, f);
    }
  }

  // ---- stage the whole weight matrix into LDS (all 256 threads) ----
#if defined(__gfx1250__)
  for (int c = threadIdx.x; c < CHUNKS; c += 256) {
    unsigned long long gaddr = (unsigned long long)(const void*)(Wt + (size_t)c * 8);
    unsigned lds_off = (unsigned)(unsigned long long)(lds_us*)(lds_w + (size_t)c * 8);
    asm volatile("global_load_async_to_lds_b128 %0, %1, off"
                 :: "v"(lds_off), "v"(gaddr) : "memory");
  }
  asm volatile("s_wait_asynccnt 0x0" ::: "memory");
#else
  for (int c = threadIdx.x; c < CHUNKS; c += 256)
    ((uint4*)lds_w)[c] = ((const uint4*)Wt)[c];
#endif
  __syncthreads();

  if (active) {
#pragma unroll
    for (int nt = 0; nt < 16; ++nt) {
      const int n0 = nt * 16;
      const float bv = bias[n0 + low];
      v8f acc = {bv, bv, bv, bv, bv, bv, bv, bv};
#pragma unroll
      for (int ks = 0; ks < KS; ++ks) {
        const unsigned short* bp = lds_w + (size_t)(n0 + low) * K + ks * 32 + half * 8;
        Frag32 f;
        f.lo = *(const uint4*)(bp);       // ds_load_b128
        f.hi = *(const uint4*)(bp + 16);  // ds_load_b128
        v16bf bfrag = __builtin_bit_cast(v16bf, f);
        acc = __builtin_amdgcn_wmma_f32_16x16x32_bf16(
            /*neg_a=*/false, afrag[ks], /*neg_b=*/false, bfrag,
            /*c_mod=*/(short)0, acc, /*reuse_a=*/false, /*reuse_b=*/false);
      }
      // C/D layout: VGPR r -> row = r + half*8, col = low (+ n0)
#pragma unroll
      for (int r = 0; r < 8; ++r) {
        float v = acc[r];
        if (RELU) v = fmaxf(v, 0.0f);
        const size_t oi = (size_t)(mtile * 16 + half * 8 + r) * HID + n0 + low;
        if (OUT_BF16) ((unsigned short*)out)[oi] = f2bf(v);
        else          ((float*)out)[oi]          = v;
      }
    }
  }
}

// ---------------------------------------------------------------------------
// BatchNorm statistics + apply, mean pool
// ---------------------------------------------------------------------------
__global__ void stats_kernel(const float* __restrict__ y, float* __restrict__ sums,
                             float* __restrict__ sqs) {
  const int c = threadIdx.x;                     // 256 threads = 256 columns
  float s = 0.f, q = 0.f;
  for (int r = blockIdx.x; r < N_NODES; r += gridDim.x) {
    float v = y[(size_t)r * HID + c];            // coalesced row reads
    s += v; q += v * v;
  }
  atomicAdd(&sums[c], s);
  atomicAdd(&sqs[c], q);
}

__global__ void finalize_stats_kernel(const float* __restrict__ sums,
                                      const float* __restrict__ sqs,
                                      float* __restrict__ mean, float* __restrict__ inv) {
  const int c = threadIdx.x;
  float m = sums[c] * (1.0f / N_NODES);
  float var = sqs[c] * (1.0f / N_NODES) - m * m;   // biased var
  mean[c] = m;
  inv[c]  = rsqrtf(var + BN_EPS);
}

__global__ void bn_relu_kernel(const float* __restrict__ y, const float* __restrict__ mean,
                               const float* __restrict__ inv, const float* __restrict__ gamma,
                               const float* __restrict__ beta, float* __restrict__ out) {
  size_t i = (size_t)blockIdx.x * blockDim.x + threadIdx.x;
  if (i >= (size_t)N_NODES * HID) return;
  int c = (int)(i & (HID - 1));
  float v = (y[i] - mean[c]) * inv[c] * gamma[c] + beta[c];
  out[i] = fmaxf(v, 0.0f);
}

__global__ void pool_kernel(const float* __restrict__ h, const int* __restrict__ batch,
                            float* __restrict__ sums, float* __restrict__ cnt) {
  size_t i = (size_t)blockIdx.x * blockDim.x + threadIdx.x;
  if (i >= (size_t)N_NODES * HID) return;
  int node = (int)(i >> 8);
  int c    = (int)(i & 255);
  int g = batch[node];
  atomicAdd(&sums[(size_t)g * HID + c], h[i]);
  if (c == 0) atomicAdd(&cnt[g], 1.0f);
}

__global__ void pool_div_kernel(const float* __restrict__ sums, const float* __restrict__ cnt,
                                float* __restrict__ out) {
  int i = blockIdx.x * blockDim.x + threadIdx.x;   // N_GRAPHS*HID
  out[i] = sums[i] / fmaxf(cnt[i >> 8], 1.0f);
}

// ---------------------------------------------------------------------------
static inline int cdiv(long long a, long long b) { return (int)((a + b - 1) / b); }

extern "C" void kernel_launch(void* const* d_in, const int* in_sizes, int n_in,
                              void* d_out, int out_size, void* d_ws, size_t ws_size,
                              hipStream_t stream) {
  (void)in_sizes; (void)n_in; (void)out_size; (void)ws_size;

  const float* x     = (const float*)d_in[0];
  const int*   ei    = (const int*)d_in[1];
  const int*   batch = (const int*)d_in[2];
  const float* W1_0  = (const float*)d_in[3];
  const float* b1_0  = (const float*)d_in[4];
  const float* W2_0  = (const float*)d_in[5];
  const float* b2_0  = (const float*)d_in[6];
  const float* g0    = (const float*)d_in[7];
  const float* be0   = (const float*)d_in[8];
  const float* W1_1  = (const float*)d_in[9];
  const float* b1_1  = (const float*)d_in[10];
  const float* W2_1  = (const float*)d_in[11];
  const float* b2_1  = (const float*)d_in[12];
  const float* g1    = (const float*)d_in[13];
  const float* be1   = (const float*)d_in[14];

  // workspace carve-out (256B aligned)
  char* ws = (char*)d_ws;
  size_t off = 0;
  auto take = [&](size_t bytes) -> void* {
    void* p = ws + off;
    off = (off + bytes + 255) & ~(size_t)255;
    return p;
  };
  float*          H    = (float*)take((size_t)N_NODES * HID * 4);   // activations
  float*          AGG  = (float*)take((size_t)N_NODES * HID * 4);   // agg / MLP out
  unsigned short* ABF  = (unsigned short*)take((size_t)N_NODES * HID * 2); // bf16 A
  unsigned short* Y1   = (unsigned short*)take((size_t)N_NODES * HID * 2); // bf16 mid
  unsigned short* W1T  = (unsigned short*)take((size_t)HID * HID * 2);
  unsigned short* W2T  = (unsigned short*)take((size_t)HID * HID * 2);
  float*          SUMS = (float*)take(HID * 4);
  float*          SQS  = (float*)take(HID * 4);
  float*          MEAN = (float*)take(HID * 4);
  float*          INV  = (float*)take(HID * 4);
  float*          PSUM = (float*)take((size_t)N_GRAPHS * HID * 4);
  float*          PCNT = (float*)take(N_GRAPHS * 4);

  const int gemm_blocks = cdiv(N_NODES / 16, 8);   // 8 waves/block, 16 rows/wave
  const int nh = N_NODES * HID;
  const int lds1 = HID * IN_DIM * 2;               // 64 KB
  const int lds2 = HID * HID * 2;                  // 128 KB (of 320 KB/WGP)

  // ---------------- Layer 0 (K = IN_DIM = 128) ----------------
  {
    const int n0 = N_NODES * IN_DIM;
    copy_kernel<<<cdiv(n0, 256), 256, 0, stream>>>(x, AGG, n0);
    scatter_kernel<IN_DIM><<<cdiv((long long)N_EDGES * IN_DIM, 256), 256, 0, stream>>>(x, ei, AGG);
    f2bf_kernel<<<cdiv(n0, 256), 256, 0, stream>>>(AGG, ABF, n0);
    wconvert_kernel<IN_DIM><<<cdiv(IN_DIM * HID, 256), 256, 0, stream>>>(W1_0, W1T);
    wconvert_kernel<HID><<<cdiv(HID * HID, 256), 256, 0, stream>>>(W2_0, W2T);
    gemm_kernel<IN_DIM, true,  true ><<<gemm_blocks, 256, lds1, stream>>>(ABF, W1T, b1_0, Y1,  N_NODES);
    gemm_kernel<HID,    false, false><<<gemm_blocks, 256, lds2, stream>>>(Y1,  W2T, b2_0, AGG, N_NODES);
    zero_kernel<<<1, 256, 0, stream>>>(SUMS, HID);
    zero_kernel<<<1, 256, 0, stream>>>(SQS,  HID);
    stats_kernel<<<512, 256, 0, stream>>>(AGG, SUMS, SQS);
    finalize_stats_kernel<<<1, 256, 0, stream>>>(SUMS, SQS, MEAN, INV);
    bn_relu_kernel<<<cdiv(nh, 256), 256, 0, stream>>>(AGG, MEAN, INV, g0, be0, H);
  }

  // ---------------- Layer 1 (K = HID = 256) ----------------
  {
    copy_kernel<<<cdiv(nh, 256), 256, 0, stream>>>(H, AGG, nh);
    scatter_kernel<HID><<<cdiv((long long)N_EDGES * HID, 256), 256, 0, stream>>>(H, ei, AGG);
    f2bf_kernel<<<cdiv(nh, 256), 256, 0, stream>>>(AGG, ABF, nh);
    wconvert_kernel<HID><<<cdiv(HID * HID, 256), 256, 0, stream>>>(W1_1, W1T);
    wconvert_kernel<HID><<<cdiv(HID * HID, 256), 256, 0, stream>>>(W2_1, W2T);
    gemm_kernel<HID, true,  true ><<<gemm_blocks, 256, lds2, stream>>>(ABF, W1T, b1_1, Y1,  N_NODES);
    gemm_kernel<HID, false, false><<<gemm_blocks, 256, lds2, stream>>>(Y1,  W2T, b2_1, AGG, N_NODES);
    zero_kernel<<<1, 256, 0, stream>>>(SUMS, HID);
    zero_kernel<<<1, 256, 0, stream>>>(SQS,  HID);
    stats_kernel<<<512, 256, 0, stream>>>(AGG, SUMS, SQS);
    finalize_stats_kernel<<<1, 256, 0, stream>>>(SUMS, SQS, MEAN, INV);
    bn_relu_kernel<<<cdiv(nh, 256), 256, 0, stream>>>(AGG, MEAN, INV, g1, be1, H);
  }

  // ---------------- global mean pool ----------------
  zero_kernel<<<cdiv(N_GRAPHS * HID, 256), 256, 0, stream>>>(PSUM, N_GRAPHS * HID);
  zero_kernel<<<1, 256, 0, stream>>>(PCNT, N_GRAPHS);
  pool_kernel<<<cdiv(nh, 256), 256, 0, stream>>>(H, batch, PSUM, PCNT);
  pool_div_kernel<<<N_GRAPHS * HID / 256, 256, 0, stream>>>(PSUM, PCNT, (float*)d_out);
}
